// GraphTransformer_38457137168645
// MI455X (gfx1250) — matrix operand order repaired and checked by reading, not compiled, not run on hardware
//
#include <hip/hip_runtime.h>
#include <hip/hip_bf16.h>

// ---------------------------------------------------------------------------
// CDNA5 (gfx1250) graph-transformer layer.
// Strategy: all GEMMs in bf16 WMMA (v_wmma_f32_16x16x32_bf16, f32 accum),
// everything else (softmax, LayerNorm, residuals, scatters) in fp32 VALU.
// ---------------------------------------------------------------------------

typedef __attribute__((ext_vector_type(16))) __bf16 v16bf;
typedef __attribute__((ext_vector_type(8)))  __bf16 v8bf;
typedef __attribute__((ext_vector_type(8)))  float  v8f;
typedef __bf16 bf16_t;

#define DD        128
#define HID       512
#define NEG_SLOPE 0.2f
#define LN_EPS    1e-5f

__device__ __forceinline__ float leaky(float x) { return x > 0.f ? x : NEG_SLOPE * x; }
__device__ __forceinline__ int   imin(int a, int b) { return a < b ? a : b; }

__device__ __forceinline__ v8f wmma_bf16(v16bf a, v16bf b, v8f c) {
  // 8 args: (neg_a, A, neg_b, B, c_mod, C, reuse_a, reuse_b)
  return __builtin_amdgcn_wmma_f32_16x16x32_bf16(false, a, false, b, (short)0, c,
                                                 false, false);
}

// A fragment (16xK row-major tile, K slice [k0,k0+32)).
// Lane L<16 owns row L and K {k0..k0+7, k0+16..k0+23}; lane L>=16 owns row
// L-16 and K {k0+8..k0+15, k0+24..k0+31}.  rowPtr = start of this lane's row.
__device__ __forceinline__ v16bf a_frag(const bf16_t* rowPtr, int k0, int lane) {
  const int klo = k0 + ((lane & 16) ? 8 : 0);
  v8bf lo = *(const v8bf*)(rowPtr + klo);
  v8bf hi = *(const v8bf*)(rowPtr + klo + 16);
  v16bf r;
#pragma unroll
  for (int i = 0; i < 8; ++i) { r[i] = lo[i]; r[i + 8] = hi[i]; }
  return r;
}

// B fragment: computing X @ W^T with torch-style W (out x in) row-major means
// B[k,n] = W[n,k].  Lane n<16 holds K {k0..k0+7, k0+16..k0+23} of column n
// (lanes 16-31 K {k0+8..k0+15, k0+24..k0+31}), matching a_frag.
__device__ __forceinline__ v16bf b_frag(const bf16_t* w, int ldk, int n0, int k0,
                                        int lane) {
  const int n  = lane & 15;
  const int kc = k0 + ((lane & 16) ? 8 : 0);
  const bf16_t* p = w + (size_t)(n0 + n) * ldk + kc;
  v8bf lo = *(const v8bf*)(p);
  v8bf hi = *(const v8bf*)(p + 16);
  v16bf r;
#pragma unroll
  for (int i = 0; i < 8; ++i) { r[i] = lo[i]; r[i + 8] = hi[i]; }
  return r;
}

// ------------------------------ utility kernels -----------------------------

__global__ __launch_bounds__(256) void cvt_bf16_kernel(const float* __restrict__ src,
                                                       bf16_t* __restrict__ dst,
                                                       long long n) {
  long long i = (long long)blockIdx.x * 256 + threadIdx.x;
  if (i < n) dst[i] = (bf16_t)src[i];
}

__global__ __launch_bounds__(256) void zero_u32_kernel(unsigned* __restrict__ p,
                                                       long long n) {
  long long i = (long long)blockIdx.x * 256 + threadIdx.x;
  if (i < n) p[i] = 0u;
}

// ------------------------- fused edge-update kernel -------------------------
// Per 16-edge tile: th/tt projections (WMMA) -> 2-way attention -> LN ->
// FFN 128->512->128 (WMMA, hidden in LDS as bf16) -> LN -> new_edge_emb.

__global__ __launch_bounds__(256) void edge_update_kernel(
    const float*  __restrict__ edge_f32,
    const bf16_t* __restrict__ node_bf,
    const int*    __restrict__ head, const int* __restrict__ tail,
    const bf16_t* __restrict__ wrs, const float* __restrict__ wrs_b,
    const bf16_t* __restrict__ wro, const float* __restrict__ wro_b,
    const float*  __restrict__ e2n_w, const float* __restrict__ e2n_b,
    const float*  __restrict__ ln1_g, const float* __restrict__ ln1_b,
    const bf16_t* __restrict__ l1w, const float* __restrict__ l1b,
    const bf16_t* __restrict__ l2w, const float* __restrict__ l2b,
    const float*  __restrict__ ln2_g, const float* __restrict__ ln2_b,
    float* __restrict__ out_edge, int E) {
  __shared__ int sHead[16], sTail[16];
  __shared__ __align__(16) float  sTh[16][DD];
  __shared__ __align__(16) float  sTt[16][DD];
  __shared__ __align__(16) float  sZ[16][DD];
  __shared__ __align__(16) bf16_t sZb[16][DD];
  __shared__ __align__(16) bf16_t sHid[16][HID];
  __shared__ float sRed[16][16];
  __shared__ float sRed2[16][16];

  const int tid  = threadIdx.x;
  const int wave = tid >> 5, lane = tid & 31;
  const int e0   = (int)blockIdx.x * 16;

  if (tid < 16) {
    const int e = imin(e0 + tid, E - 1);
    sHead[tid] = head[e];
    sTail[tid] = tail[e];
  }
  __syncthreads();

  // ---- th = node[head] @ Wrs^T + b ; tt = node[tail] @ Wro^T + b ----
  {
    const int n0 = wave * 16;
    const int m  = lane & 15;
    const bf16_t* ah = node_bf + (size_t)sHead[m] * DD;
    const bf16_t* at = node_bf + (size_t)sTail[m] * DD;
    v8f cth = {}; v8f ctt = {};
#pragma unroll
    for (int k0 = 0; k0 < DD; k0 += 32) {
      cth = wmma_bf16(a_frag(ah, k0, lane), b_frag(wrs, DD, n0, k0, lane), cth);
      ctt = wmma_bf16(a_frag(at, k0, lane), b_frag(wro, DD, n0, k0, lane), ctt);
    }
    const int col = n0 + (lane & 15);
    const int rb  = (lane & 16) ? 8 : 0;
    const float bh = wrs_b[col], bt = wro_b[col];
#pragma unroll
    for (int r = 0; r < 8; ++r) {
      sTh[rb + r][col] = cth[r] + bh;
      sTt[rb + r][col] = ctt[r] + bt;
    }
  }
  __syncthreads();

  // ---- attention logits ha/ta (row-wise, 16 threads per row) ----
  const int row  = tid >> 4, part = tid & 15;
  const int eG   = imin(e0 + row, E - 1);
  const float* erow = edge_f32 + (size_t)eG * DD;
  {
    float pe = 0.f, pth = 0.f, ptt = 0.f;
#pragma unroll
    for (int j = 0; j < 8; ++j) {
      const int k = part * 8 + j;
      const float ev = erow[k];
      pe  += ev * e2n_w[k];
      pth += sTh[row][k] * e2n_w[DD + k];
      ptt += sTt[row][k] * e2n_w[DD + k];
    }
    sRed[row][part]  = pe + pth;
    sRed2[row][part] = pe + ptt;
  }
  __syncthreads();
  float att0;
  {
    float ha = 0.f, ta = 0.f;
#pragma unroll
    for (int i = 0; i < 16; ++i) { ha += sRed[row][i]; ta += sRed2[row][i]; }
    ha = leaky(ha + e2n_b[0]);
    ta = leaky(ta + e2n_b[0]);
    att0 = 1.f / (1.f + __expf(ta - ha));   // softmax over {ha,ta}
  }
  __syncthreads();

  // ---- new_feat + residual, then LayerNorm 1 ----
  float v[8];
  {
    float s1 = 0.f, s2 = 0.f;
#pragma unroll
    for (int j = 0; j < 8; ++j) {
      const int k = part * 8 + j;
      const float x = att0 * sTh[row][k] + (1.f - att0) * sTt[row][k] + erow[k];
      v[j] = x; s1 += x; s2 += x * x;
    }
    sRed[row][part] = s1; sRed2[row][part] = s2;
  }
  __syncthreads();
  {
    float s1 = 0.f, s2 = 0.f;
#pragma unroll
    for (int i = 0; i < 16; ++i) { s1 += sRed[row][i]; s2 += sRed2[row][i]; }
    const float mu  = s1 * (1.f / DD);
    const float var = s2 * (1.f / DD) - mu * mu;
    const float rs  = rsqrtf(var + LN_EPS);
#pragma unroll
    for (int j = 0; j < 8; ++j) {
      const int k = part * 8 + j;
      const float z = (v[j] - mu) * rs * ln1_g[k] + ln1_b[k];
      sZ[row][k]  = z;
      sZb[row][k] = (bf16_t)z;
    }
  }
  __syncthreads();

  // ---- FFN layer 1: 128 -> 512, hidden kept as bf16 in LDS ----
  {
    const bf16_t* arow = &sZb[lane & 15][0];
#pragma unroll
    for (int t = 0; t < 4; ++t) {
      const int n0 = (wave * 4 + t) * 16;
      v8f c = {};
#pragma unroll
      for (int k0 = 0; k0 < DD; k0 += 32)
        c = wmma_bf16(a_frag(arow, k0, lane), b_frag(l1w, DD, n0, k0, lane), c);
      const int col = n0 + (lane & 15);
      const int rb  = (lane & 16) ? 8 : 0;
      const float bv = l1b[col];
#pragma unroll
      for (int r = 0; r < 8; ++r)
        sHid[rb + r][col] = (bf16_t)leaky(c[r] + bv);
    }
  }
  __syncthreads();

  // ---- FFN layer 2: 512 -> 128, + residual (edge_z) ----
  {
    const bf16_t* arow = &sHid[lane & 15][0];
    const int n0 = wave * 16;
    v8f c = {};
#pragma unroll
    for (int k0 = 0; k0 < HID; k0 += 32)
      c = wmma_bf16(a_frag(arow, k0, lane), b_frag(l2w, HID, n0, k0, lane), c);
    const int col = n0 + (lane & 15);
    const int rb  = (lane & 16) ? 8 : 0;
    const float bv = l2b[col];
#pragma unroll
    for (int r = 0; r < 8; ++r)
      sTh[rb + r][col] = leaky(c[r] + bv) + sZ[rb + r][col];  // reuse sTh
  }
  __syncthreads();

  // ---- LayerNorm 2 + store new_edge_emb ----
  {
    float w[8]; float s1 = 0.f, s2 = 0.f;
#pragma unroll
    for (int j = 0; j < 8; ++j) {
      const int k = part * 8 + j;
      const float x = sTh[row][k];
      w[j] = x; s1 += x; s2 += x * x;
    }
    sRed[row][part] = s1; sRed2[row][part] = s2;
    __syncthreads();
    s1 = 0.f; s2 = 0.f;
#pragma unroll
    for (int i = 0; i < 16; ++i) { s1 += sRed[row][i]; s2 += sRed2[row][i]; }
    const float mu  = s1 * (1.f / DD);
    const float var = s2 * (1.f / DD) - mu * mu;
    const float rs  = rsqrtf(var + LN_EPS);
    if (e0 + row < E) {
      float* orow = out_edge + (size_t)(e0 + row) * DD;
#pragma unroll
      for (int j = 0; j < 8; ++j) {
        const int k = part * 8 + j;
        orow[k] = (w[j] - mu) * rs * ln2_g[k] + ln2_b[k];
      }
    }
  }
}

// ---------------- edge->node: ef GEMM + attention logits + seg-max ----------

__global__ __launch_bounds__(256) void ef_att_kernel(
    const bf16_t* __restrict__ edge_bf,
    const float*  __restrict__ node_f32,
    const int*    __restrict__ idx,
    const bf16_t* __restrict__ w, const float* __restrict__ wb,
    const float*  __restrict__ n2e_w, const float* __restrict__ n2e_b,
    float* __restrict__ ef_out, float* __restrict__ a_out,
    unsigned* __restrict__ segm, int E) {
  __shared__ __align__(16) float sEf[16][DD];
  __shared__ int sIdx[16];
  const int tid  = threadIdx.x;
  const int wave = tid >> 5, lane = tid & 31;
  const int e0   = (int)blockIdx.x * 16;
  if (tid < 16) sIdx[tid] = idx[imin(e0 + tid, E - 1)];

  // ef = edge_emb @ W^T + b  (WMMA)
  {
    const int m = lane & 15;
    const bf16_t* arow = edge_bf + (size_t)imin(e0 + m, E - 1) * DD;
    const int n0 = wave * 16;
    v8f c = {};
#pragma unroll
    for (int k0 = 0; k0 < DD; k0 += 32)
      c = wmma_bf16(a_frag(arow, k0, lane), b_frag(w, DD, n0, k0, lane), c);
    const int col = n0 + (lane & 15);
    const int rb  = (lane & 16) ? 8 : 0;
    const float bv = wb[col];
#pragma unroll
    for (int r = 0; r < 8; ++r) {
      const float val = c[r] + bv;
      sEf[rb + r][col] = val;
      const int e = e0 + rb + r;
      if (e < E) ef_out[(size_t)e * DD + col] = val;
    }
  }
  __syncthreads();

  // a[e,h] = leaky(concat(node[idx], ef) . n2e_w[h] + b[h]); seg-max via
  // order-preserving uint encoding + atomicMax.
  if (tid < 128) {
    const int r = tid >> 3, h = tid & 7;
    const int e = e0 + r;
    if (e < E) {
      const int node = sIdx[r];
      const float* hn = node_f32 + (size_t)node * DD;
      const float* w0 = n2e_w + (size_t)h * (2 * DD);
      float acc = n2e_b[h];
      for (int k = 0; k < DD; ++k) acc += hn[k] * w0[k];
      for (int k = 0; k < DD; ++k) acc += sEf[r][k] * w0[DD + k];
      acc = leaky(acc);
      a_out[(size_t)e * 8 + h] = acc;
      const unsigned u   = __float_as_uint(acc);
      const unsigned enc = (u & 0x80000000u) ? ~u : (u | 0x80000000u);
      atomicMax(&segm[(size_t)node * 8 + h], enc);
    }
  }
}

__global__ __launch_bounds__(256) void seg_expsum_kernel(
    const int* __restrict__ idx, float* __restrict__ a_out,
    const unsigned* __restrict__ segm, float* __restrict__ segs, int E) {
  const long long i = (long long)blockIdx.x * 256 + threadIdx.x;
  if (i < (long long)E * 8) {
    const int e = (int)(i >> 3), h = (int)(i & 7);
    const int node = idx[e];
    const unsigned u = segm[(size_t)node * 8 + h];
    const float m = (u & 0x80000000u) ? __uint_as_float(u ^ 0x80000000u)
                                      : __uint_as_float(~u);
    const float ex = __expf(a_out[i] - m);
    a_out[i] = ex;
    atomicAdd(&segs[(size_t)node * 8 + h], ex);
  }
}

__global__ __launch_bounds__(256) void scatter_msg_kernel(
    const int* __restrict__ idx, const float* __restrict__ a_ex,
    const float* __restrict__ segs, const float* __restrict__ ef,
    float* __restrict__ acc, int E) {
  __shared__ float sCoef[2];
  const int tid = threadIdx.x;
  const int sub = tid >> 7;
  const int c   = tid & 127;
  const int e   = (int)blockIdx.x * 2 + sub;
  int node = 0;
  if (e < E) {
    node = idx[e];
    if (c == 0) {
      float coef = 0.f;
#pragma unroll
      for (int h = 0; h < 8; ++h)
        coef += a_ex[(size_t)e * 8 + h] / (segs[(size_t)node * 8 + h] + 1e-16f);
      sCoef[sub] = coef * 0.125f;  // / H
    }
  }
  __syncthreads();
  if (e < E) {
    const float coef = sCoef[sub];
    atomicAdd(&acc[(size_t)node * DD + c], coef * ef[(size_t)e * DD + c]);
  }
}

// ---------------- node_z = LN(acc + node_emb), f32 + bf16 copies ------------

__global__ __launch_bounds__(256) void node_z_kernel(
    const float* __restrict__ acc, const float* __restrict__ node_f32,
    const float* __restrict__ g, const float* __restrict__ b,
    float* __restrict__ z_f32, bf16_t* __restrict__ z_bf, int N) {
  __shared__ float sRed[16][16], sRed2[16][16];
  const int tid = threadIdx.x;
  const int row = tid >> 4, part = tid & 15;
  const int n   = (int)blockIdx.x * 16 + row;
  const int nC  = imin(n, N - 1);
  const float* ar = acc + (size_t)nC * DD;
  const float* xr = node_f32 + (size_t)nC * DD;
  float v[8]; float s1 = 0.f, s2 = 0.f;
#pragma unroll
  for (int j = 0; j < 8; ++j) {
    const int k = part * 8 + j;
    const float x = ar[k] + xr[k];
    v[j] = x; s1 += x; s2 += x * x;
  }
  sRed[row][part] = s1; sRed2[row][part] = s2;
  __syncthreads();
  s1 = 0.f; s2 = 0.f;
#pragma unroll
  for (int i = 0; i < 16; ++i) { s1 += sRed[row][i]; s2 += sRed2[row][i]; }
  const float mu  = s1 * (1.f / DD);
  const float var = s2 * (1.f / DD) - mu * mu;
  const float rs  = rsqrtf(var + LN_EPS);
  if (n < N) {
#pragma unroll
    for (int j = 0; j < 8; ++j) {
      const int k = part * 8 + j;
      const float z = (v[j] - mu) * rs * g[k] + b[k];
      z_f32[(size_t)n * DD + k] = z;
      z_bf[(size_t)n * DD + k]  = (bf16_t)z;
    }
  }
}

// ---------------- fused node FFN + LN -> new_node_emb -----------------------

__global__ __launch_bounds__(256) void node_ffn_kernel(
    const float*  __restrict__ z_f32, const bf16_t* __restrict__ z_bf,
    const bf16_t* __restrict__ l1w, const float* __restrict__ l1b,
    const bf16_t* __restrict__ l2w, const float* __restrict__ l2b,
    const float*  __restrict__ ln_g, const float* __restrict__ ln_b,
    float* __restrict__ out, int N) {
  __shared__ __align__(16) bf16_t sHid[16][HID];
  __shared__ __align__(16) float  sOut[16][DD];
  __shared__ float sRed[16][16], sRed2[16][16];
  const int tid  = threadIdx.x;
  const int wave = tid >> 5, lane = tid & 31;
  const int r0   = (int)blockIdx.x * 16;

  // FFN1: 128 -> 512
  {
    const bf16_t* arow = z_bf + (size_t)imin(r0 + (lane & 15), N - 1) * DD;
#pragma unroll
    for (int t = 0; t < 4; ++t) {
      const int n0 = (wave * 4 + t) * 16;
      v8f c = {};
#pragma unroll
      for (int k0 = 0; k0 < DD; k0 += 32)
        c = wmma_bf16(a_frag(arow, k0, lane), b_frag(l1w, DD, n0, k0, lane), c);
      const int col = n0 + (lane & 15);
      const int rb  = (lane & 16) ? 8 : 0;
      const float bv = l1b[col];
#pragma unroll
      for (int r = 0; r < 8; ++r)
        sHid[rb + r][col] = (bf16_t)leaky(c[r] + bv);
    }
  }
  __syncthreads();

  // FFN2: 512 -> 128, + residual
  {
    const bf16_t* arow = &sHid[lane & 15][0];
    const int n0 = wave * 16;
    v8f c = {};
#pragma unroll
    for (int k0 = 0; k0 < HID; k0 += 32)
      c = wmma_bf16(a_frag(arow, k0, lane), b_frag(l2w, HID, n0, k0, lane), c);
    const int col = n0 + (lane & 15);
    const int rb  = (lane & 16) ? 8 : 0;
    const float bv = l2b[col];
#pragma unroll
    for (int r = 0; r < 8; ++r) {
      const int gr = imin(r0 + rb + r, N - 1);
      sOut[rb + r][col] = leaky(c[r] + bv) + z_f32[(size_t)gr * DD + col];
    }
  }
  __syncthreads();

  // LayerNorm + store
  {
    const int row = tid >> 4, part = tid & 15;
    float w[8]; float s1 = 0.f, s2 = 0.f;
#pragma unroll
    for (int j = 0; j < 8; ++j) {
      const int k = part * 8 + j;
      const float x = sOut[row][k];
      w[j] = x; s1 += x; s2 += x * x;
    }
    sRed[row][part] = s1; sRed2[row][part] = s2;
    __syncthreads();
    s1 = 0.f; s2 = 0.f;
#pragma unroll
    for (int i = 0; i < 16; ++i) { s1 += sRed[row][i]; s2 += sRed2[row][i]; }
    const float mu  = s1 * (1.f / DD);
    const float var = s2 * (1.f / DD) - mu * mu;
    const float rs  = rsqrtf(var + LN_EPS);
    const int n = r0 + row;
    if (n < N) {
      float* orow = out + (size_t)n * DD;
#pragma unroll
      for (int j = 0; j < 8; ++j) {
        const int k = part * 8 + j;
        orow[k] = (w[j] - mu) * rs * ln_g[k] + ln_b[k];
      }
    }
  }
}

// ----------------------------------------------------------------------------

extern "C" void kernel_launch(void* const* d_in, const int* in_sizes, int n_in,
                              void* d_out, int out_size, void* d_ws, size_t ws_size,
                              hipStream_t stream) {
  (void)n_in; (void)out_size; (void)ws_size;
  const float* node_emb = (const float*)d_in[0];
  const float* edge_emb = (const float*)d_in[1];
  const int*   head     = (const int*)d_in[2];
  const int*   tail     = (const int*)d_in[3];
  const float* Wsr_w = (const float*)d_in[4];  const float* Wsr_b = (const float*)d_in[5];
  const float* Wor_w = (const float*)d_in[6];  const float* Wor_b = (const float*)d_in[7];
  const float* Wrs_w = (const float*)d_in[8];  const float* Wrs_b = (const float*)d_in[9];
  const float* Wro_w = (const float*)d_in[10]; const float* Wro_b = (const float*)d_in[11];
  const float* n2e_w = (const float*)d_in[12]; const float* n2e_b = (const float*)d_in[13];
  const float* e2n_w = (const float*)d_in[14]; const float* e2n_b = (const float*)d_in[15];
  const float* e_ln1_g = (const float*)d_in[16]; const float* e_ln1_b = (const float*)d_in[17];
  const float* e_ln2_g = (const float*)d_in[18]; const float* e_ln2_b = (const float*)d_in[19];
  const float* n_ln1_g = (const float*)d_in[20]; const float* n_ln1_b = (const float*)d_in[21];
  const float* n_ln2_g = (const float*)d_in[22]; const float* n_ln2_b = (const float*)d_in[23];
  const float* e_l1_w = (const float*)d_in[24]; const float* e_l1_b = (const float*)d_in[25];
  const float* e_l2_w = (const float*)d_in[26]; const float* e_l2_b = (const float*)d_in[27];
  const float* n_l1_w = (const float*)d_in[28]; const float* n_l1_b = (const float*)d_in[29];
  const float* n_l2_w = (const float*)d_in[30]; const float* n_l2_b = (const float*)d_in[31];

  const int N = in_sizes[0] / DD;
  const int E = in_sizes[2];

  // -------- workspace carve (256B aligned regions) --------
  char* wsp = (char*)d_ws;
  size_t off = 0;
  auto carve = [&](size_t bytes) -> void* {
    void* p = wsp + off;
    off += (bytes + 255) & ~(size_t)255;
    return p;
  };
  bf16_t* node_bf = (bf16_t*)carve((size_t)N * DD * 2);
  bf16_t* edge_bf = (bf16_t*)carve((size_t)E * DD * 2);
  bf16_t* wrs_bf  = (bf16_t*)carve((size_t)DD * DD * 2);
  bf16_t* wro_bf  = (bf16_t*)carve((size_t)DD * DD * 2);
  bf16_t* wsr_bf  = (bf16_t*)carve((size_t)DD * DD * 2);
  bf16_t* wor_bf  = (bf16_t*)carve((size_t)DD * DD * 2);
  bf16_t* el1_bf  = (bf16_t*)carve((size_t)HID * DD * 2);
  bf16_t* el2_bf  = (bf16_t*)carve((size_t)DD * HID * 2);
  bf16_t* nl1_bf  = (bf16_t*)carve((size_t)HID * DD * 2);
  bf16_t* nl2_bf  = (bf16_t*)carve((size_t)DD * HID * 2);
  float*    ef    = (float*)carve((size_t)E * DD * 4);
  float*    a_e   = (float*)carve((size_t)E * 8 * 4);
  unsigned* segm  = (unsigned*)carve((size_t)N * 8 * 4);
  float*    segs  = (float*)carve((size_t)N * 8 * 4);
  float*    acc   = (float*)carve((size_t)N * DD * 4);
  float*    z_f32 = (float*)carve((size_t)N * DD * 4);
  bf16_t*   z_bf  = (bf16_t*)carve((size_t)N * DD * 2);

  float* out_node = (float*)d_out;
  float* out_edge = out_node + (size_t)N * DD;

  auto cvt = [&](const float* s, bf16_t* d, long long n) {
    cvt_bf16_kernel<<<dim3((unsigned)((n + 255) / 256)), dim3(256), 0, stream>>>(s, d, n);
  };
  auto zero = [&](void* p, long long n32) {
    zero_u32_kernel<<<dim3((unsigned)((n32 + 255) / 256)), dim3(256), 0, stream>>>(
        (unsigned*)p, n32);
  };

  // -------- bf16 conversions --------
  cvt(node_emb, node_bf, (long long)N * DD);
  cvt(edge_emb, edge_bf, (long long)E * DD);
  cvt(Wrs_w, wrs_bf, DD * DD);
  cvt(Wro_w, wro_bf, DD * DD);
  cvt(Wsr_w, wsr_bf, DD * DD);
  cvt(Wor_w, wor_bf, DD * DD);
  cvt(e_l1_w, el1_bf, HID * DD);
  cvt(e_l2_w, el2_bf, DD * HID);
  cvt(n_l1_w, nl1_bf, HID * DD);
  cvt(n_l2_w, nl2_bf, DD * HID);

  const int eTiles = (E + 15) / 16;
  const int nTiles = (N + 15) / 16;

  // -------- edge pipeline (fully fused) --------
  edge_update_kernel<<<dim3(eTiles), dim3(256), 0, stream>>>(
      edge_emb, node_bf, head, tail, wrs_bf, Wrs_b, wro_bf, Wro_b, e2n_w, e2n_b,
      e_ln1_g, e_ln1_b, el1_bf, e_l1_b, el2_bf, e_l2_b, e_ln2_g, e_ln2_b,
      out_edge, E);

  // -------- edge -> node aggregation (head then tail into one accumulator) --
  zero(acc, (long long)N * DD);
  for (int dir = 0; dir < 2; ++dir) {
    const int*    idx = dir ? tail : head;
    const bf16_t* w   = dir ? wor_bf : wsr_bf;
    const float*  wb  = dir ? Wor_b : Wsr_b;
    zero(segm, (long long)N * 8);
    zero(segs, (long long)N * 8);
    ef_att_kernel<<<dim3(eTiles), dim3(256), 0, stream>>>(
        edge_bf, node_emb, idx, w, wb, n2e_w, n2e_b, ef, a_e, segm, E);
    seg_expsum_kernel<<<dim3((unsigned)(((long long)E * 8 + 255) / 256)), dim3(256), 0,
                        stream>>>(idx, a_e, segm, segs, E);
    scatter_msg_kernel<<<dim3((E + 1) / 2), dim3(256), 0, stream>>>(idx, a_e, segs,
                                                                    ef, acc, E);
  }

  // -------- node LN + FFN --------
  node_z_kernel<<<dim3(nTiles), dim3(256), 0, stream>>>(acc, node_emb, n_ln1_g,
                                                        n_ln1_b, z_f32, z_bf, N);
  node_ffn_kernel<<<dim3(nTiles), dim3(256), 0, stream>>>(
      z_f32, z_bf, nl1_bf, n_l1_b, nl2_bf, n_l2_b, n_ln2_g, n_ln2_b, out_node, N);
}
